// CostVolumeDepthPredictor_79448305041759
// MI455X (gfx1250) — matrix operand order repaired and checked by reading, not compiled, not run on hardware
//
#include <hip/hip_runtime.h>
#include <hip/hip_bf16.h>
#include <math.h>

typedef __attribute__((ext_vector_type(16))) _Float16 v16h;
typedef __attribute__((ext_vector_type(8)))  _Float16 v8h;
typedef __attribute__((ext_vector_type(8)))  float    v8f;

#define V_  4
#define C_  64
#define D_  64
#define LH_ 64
#define LW_ 80
#define FH_ 256
#define FW_ 320
#define IH_ 512
#define IW_ 640
#define KFR_ 603          // 9 taps * 67 channels
#define KFRP_ 608         // padded to multiple of 32

__device__ __forceinline__ int imin(int a, int b) { return a < b ? a : b; }
__device__ __forceinline__ int imax(int a, int b) { return a > b ? a : b; }
__device__ __forceinline__ float gelu_exact(float x) {
    return 0.5f * x * (1.f + erff(x * 0.70710678118654752f));
}
__device__ __forceinline__ v16h cat16(v8h lo, v8h hi) {
    return __builtin_shufflevector(lo, hi, 0,1,2,3,4,5,6,7,8,9,10,11,12,13,14,15);
}

// ---------------------------------------------------------------------------
// prep kernels
// ---------------------------------------------------------------------------
__global__ void cvd_f32_to_f16(const float* __restrict__ X, _Float16* __restrict__ Y, int n) {
    int i = blockIdx.x * blockDim.x + threadIdx.x;
    if (i < n) Y[i] = (_Float16)X[i];
}

// lf (V,64,Nl) -> n-major f16 (V,Nl,64); c-fastest threads => coalesced stores
__global__ void cvd_lf_to_f16t(const float* __restrict__ lf, _Float16* __restrict__ out) {
    const int Nl = LH_ * LW_;
    int idx = blockIdx.x * blockDim.x + threadIdx.x;
    int total = V_ * Nl * C_;
    if (idx >= total) return;
    int c = idx % C_;
    int t = idx / C_;
    int p = t % Nl;
    int v = t / Nl;
    out[idx] = (_Float16)lf[((size_t)v * C_ + c) * Nl + p];
}

// repack fr_conv weights (64,67,3,3) -> f16 [64][608], K = tap*67+ci, zero pad
__global__ void cvd_pack_frw(const float* __restrict__ W, _Float16* __restrict__ Wp) {
    int idx = blockIdx.x * blockDim.x + threadIdx.x;
    if (idx >= C_ * KFRP_) return;
    int m = idx / KFRP_, k = idx % KFRP_;
    float v = 0.f;
    if (k < KFR_) {
        int t = k / 67, ci = k % 67;
        v = W[(m * 67 + ci) * 9 + t];
    }
    Wp[idx] = (_Float16)v;
}

// ---------------------------------------------------------------------------
// per-view matrices: Ks, Kinv = inv(Ks), w2c = inv(E) (E rigid)
// ---------------------------------------------------------------------------
__global__ void cvd_setup_mats(const float* __restrict__ K, const float* __restrict__ E,
                               float* __restrict__ Ks, float* __restrict__ Kinv,
                               float* __restrict__ w2c) {
    int i = threadIdx.x;
    if (i >= V_) return;
    float k[9];
    const float sc[3] = {(float)LW_, (float)LH_, 1.f};
    for (int r = 0; r < 3; ++r)
        for (int c = 0; c < 3; ++c) k[r * 3 + c] = K[i * 9 + r * 3 + c] * sc[r];
    for (int j = 0; j < 9; ++j) Ks[i * 9 + j] = k[j];
    float det = k[0] * (k[4] * k[8] - k[5] * k[7])
              - k[1] * (k[3] * k[8] - k[5] * k[6])
              + k[2] * (k[3] * k[7] - k[4] * k[6]);
    float id = 1.f / det;
    float inv[9] = {
        (k[4]*k[8]-k[5]*k[7])*id, (k[2]*k[7]-k[1]*k[8])*id, (k[1]*k[5]-k[2]*k[4])*id,
        (k[5]*k[6]-k[3]*k[8])*id, (k[0]*k[8]-k[2]*k[6])*id, (k[2]*k[3]-k[0]*k[5])*id,
        (k[3]*k[7]-k[4]*k[6])*id, (k[1]*k[6]-k[0]*k[7])*id, (k[0]*k[4]-k[1]*k[3])*id};
    for (int j = 0; j < 9; ++j) Kinv[i * 9 + j] = inv[j];
    float R[9], t[3], w[16];
    for (int r = 0; r < 3; ++r) {
        for (int c = 0; c < 3; ++c) R[r * 3 + c] = E[i * 16 + r * 4 + c];
        t[r] = E[i * 16 + r * 4 + 3];
    }
    for (int r = 0; r < 3; ++r) {
        for (int c = 0; c < 3; ++c) w[r * 4 + c] = R[c * 3 + r];
        w[r * 4 + 3] = -(R[0 * 3 + r] * t[0] + R[1 * 3 + r] * t[1] + R[2 * 3 + r] * t[2]);
    }
    w[12] = 0.f; w[13] = 0.f; w[14] = 0.f; w[15] = 1.f;
    for (int j = 0; j < 16; ++j) w2c[i * 16 + j] = w[j];
}

// ---------------------------------------------------------------------------
// generic align-corners bilinear resize over P planes
// ---------------------------------------------------------------------------
__global__ void cvd_resize_bilinear(const float* __restrict__ in, float* __restrict__ out,
                                    int P, int h, int w, int oh, int ow) {
    int idx = blockIdx.x * blockDim.x + threadIdx.x;
    int total = P * oh * ow;
    if (idx >= total) return;
    int pl = idx / (oh * ow);
    int q  = idx % (oh * ow);
    int oy = q / ow, ox = q % ow;
    float sy = (float)(h - 1) / (float)(oh - 1);
    float sx = (float)(w - 1) / (float)(ow - 1);
    float fy = oy * sy, fx = ox * sx;
    int y0 = (int)floorf(fy), x0 = (int)floorf(fx);
    int y1 = imin(y0 + 1, h - 1), x1 = imin(x0 + 1, w - 1);
    float wy = fy - (float)y0, wx = fx - (float)x0;
    const float* pin = in + (size_t)pl * h * w;
    float top = pin[y0 * w + x0] * (1.f - wx) + pin[y0 * w + x1] * wx;
    float bot = pin[y1 * w + x0] * (1.f - wx) + pin[y1 * w + x1] * wx;
    out[idx] = top * (1.f - wy) + bot * wy;
}

// ---------------------------------------------------------------------------
// plane-sweep cost volume, aggregated over source views -> agg[D][LH][LW]
// ---------------------------------------------------------------------------
__global__ void cvd_cost_volume(const float* __restrict__ lf, const float* __restrict__ E,
                                const float* __restrict__ Ks, const float* __restrict__ Kinv,
                                const float* __restrict__ w2c, float* __restrict__ agg, int r) {
    const int Npx = LH_ * LW_;
    int idx = blockIdx.x * blockDim.x + threadIdx.x;
    if (idx >= D_ * Npx) return;
    int d  = idx / Npx;
    int pp = idx % Npx;
    int y = pp / LW_, x = pp % LW_;

    const float* Ki = Kinv + r * 9;
    float gx = (float)x, gy = (float)y;
    float rx = Ki[0] * gx + Ki[1] * gy + Ki[2];
    float ry = Ki[3] * gx + Ki[4] * gy + Ki[5];
    float rz = Ki[6] * gx + Ki[7] * gy + Ki[8];
    float ib = 0.05f + (float)d * (1.95f / 63.f);
    float depth = 1.f / ib;
    float px_ = rx * depth, py_ = ry * depth, pz_ = rz * depth;
    const float* Er = E + r * 16;
    float wx_ = Er[0] * px_ + Er[1] * py_ + Er[2]  * pz_ + Er[3];
    float wy_ = Er[4] * px_ + Er[5] * py_ + Er[6]  * pz_ + Er[7];
    float wz_ = Er[8] * px_ + Er[9] * py_ + Er[10] * pz_ + Er[11];

    const float* refp = lf + ((size_t)r * C_) * Npx + pp;
    float rn2 = 0.f;
    for (int c = 0; c < C_; ++c) { float v = refp[c * Npx]; rn2 += v * v; }
    float ref_n = fmaxf(sqrtf(rn2), 1e-8f);

    float sum_w = 0.f, sum_wc = 0.f;
    for (int s = 0; s < V_; ++s) {
        if (s == r) continue;
        const float* Wc = w2c + s * 16;
        float cx = Wc[0] * wx_ + Wc[1] * wy_ + Wc[2]  * wz_ + Wc[3];
        float cy = Wc[4] * wx_ + Wc[5] * wy_ + Wc[6]  * wz_ + Wc[7];
        float cz = Wc[8] * wx_ + Wc[9] * wy_ + Wc[10] * wz_ + Wc[11];
        const float* Kss = Ks + s * 9;
        float p0 = Kss[0] * cx + Kss[1] * cy + Kss[2] * cz;
        float p1 = Kss[3] * cx + Kss[4] * cy + Kss[5] * cz;
        float p2 = Kss[6] * cx + Kss[7] * cy + Kss[8] * cz;
        float zc = fmaxf(p2, 1e-6f);
        float xs = p0 / zc, ys = p1 / zc;
        float vis = (p2 > 1e-6f && xs >= 0.f && xs <= (float)(LW_ - 1) &&
                     ys >= 0.f && ys <= (float)(LH_ - 1)) ? 1.f : 0.f;
        float x0f = floorf(xs), y0f = floorf(ys);
        float fx = xs - x0f, fy = ys - y0f;
        int x0 = (int)x0f, y0 = (int)y0f;
        int x1 = x0 + 1, y1 = y0 + 1;
        float m00 = (x0 >= 0 && x0 < LW_ && y0 >= 0 && y0 < LH_) ? 1.f : 0.f;
        float m01 = (x1 >= 0 && x1 < LW_ && y0 >= 0 && y0 < LH_) ? 1.f : 0.f;
        float m10 = (x0 >= 0 && x0 < LW_ && y1 >= 0 && y1 < LH_) ? 1.f : 0.f;
        float m11 = (x1 >= 0 && x1 < LW_ && y1 >= 0 && y1 < LH_) ? 1.f : 0.f;
        int cx0 = imin(imax(x0, 0), LW_ - 1), cx1 = imin(imax(x1, 0), LW_ - 1);
        int cy0 = imin(imax(y0, 0), LH_ - 1), cy1 = imin(imax(y1, 0), LH_ - 1);
        int i00 = cy0 * LW_ + cx0, i01 = cy0 * LW_ + cx1;
        int i10 = cy1 * LW_ + cx0, i11 = cy1 * LW_ + cx1;
        float w00 = (1.f - fy) * (1.f - fx) * m00 * vis;
        float w01 = (1.f - fy) * fx        * m01 * vis;
        float w10 = fy        * (1.f - fx) * m10 * vis;
        float w11 = fy        * fx         * m11 * vis;
        const float* sp = lf + ((size_t)s * C_) * Npx;
        float dot = 0.f, n2 = 0.f;
        for (int c = 0; c < C_; ++c) {
            const float* cp = sp + c * Npx;
            float v = w00 * cp[i00] + w01 * cp[i01] + w10 * cp[i10] + w11 * cp[i11];
            dot += refp[c * Npx] * v;
            n2  += v * v;
        }
        float wn  = fmaxf(sqrtf(n2), 1e-8f);
        float cosv = dot / (ref_n * wn);
        float cost = (1.f - cosv) * vis + (1.f - vis);
        float wt = vis + 0.2f;
        sum_w  += wt;
        sum_wc += wt * cost;
    }
    agg[idx] = (sum_wc / fmaxf(sum_w, 1e-6f)) / 0.35f;
}

// ---------------------------------------------------------------------------
// depthwise 3x3 conv, pad=1, writes **n-major** f16 Yt[pixel][Cn]
// c-fastest thread mapping -> coalesced 2-byte stores
// ---------------------------------------------------------------------------
__global__ void cvd_dwconv3_f16t(const float* __restrict__ X, const float* __restrict__ Wd,
                                 _Float16* __restrict__ Yt, int Cn, int H, int Wi) {
    int idx = blockIdx.x * blockDim.x + threadIdx.x;
    int total = Cn * H * Wi;
    if (idx >= total) return;
    int c = idx % Cn;
    int p = idx / Cn;
    int y = p / Wi, x = p % Wi;
    const float* plane = X + (size_t)c * H * Wi;
    float s = 0.f;
    #pragma unroll
    for (int ky = 0; ky < 3; ++ky) {
        int yy = y + ky - 1;
        if (yy < 0 || yy >= H) continue;
        #pragma unroll
        for (int kx = 0; kx < 3; ++kx) {
            int xx = x + kx - 1;
            if (xx < 0 || xx >= Wi) continue;
            s += Wd[c * 9 + ky * 3 + kx] * plane[yy * Wi + xx];
        }
    }
    Yt[(size_t)p * Cn + c] = (_Float16)s;
}

// ---------------------------------------------------------------------------
// 1x1 conv WMMA GEMM: Y[64,N] = Wh[64,64] @ Xt[N,64]^T + bias
// A and B fragments are each two aligned v8h (b128) loads -> branch-free,
// no scalar gathers. Wave tile 16m x 64n (4 accumulators); block 64m x 128n.
// ---------------------------------------------------------------------------
__global__ void cvd_pw_gemm_wmma(const _Float16* __restrict__ Xt, const _Float16* __restrict__ Wh,
                                 const float* __restrict__ bias, float* __restrict__ Y, int N) {
    int wave = threadIdx.x >> 5;
    int lane = threadIdx.x & 31;
    int m0 = (wave & 3) * 16;
    int nb = blockIdx.x * 128 + (wave >> 2) * 64;
    int lm = lane & 15, half = lane >> 4;
    const _Float16* wrow = Wh + (m0 + lm) * 64;
    v8f acc[4] = {{}, {}, {}, {}};
    #pragma unroll
    for (int kk = 0; kk < 64; kk += 32) {
        v16h a = cat16(*reinterpret_cast<const v8h*>(wrow + kk + half * 8),
                       *reinterpret_cast<const v8h*>(wrow + kk + 16 + half * 8));
        #pragma unroll
        for (int j = 0; j < 4; ++j) {
            int n = nb + j * 16 + lm;
            const _Float16* xrow = Xt + (size_t)n * 64 + kk;
            v16h b = cat16(*reinterpret_cast<const v8h*>(xrow + half * 8),
                           *reinterpret_cast<const v8h*>(xrow + 16 + half * 8));
            acc[j] = __builtin_amdgcn_wmma_f32_16x16x32_f16(false, a, false, b, (short)0,
                                                            acc[j], false, false);
        }
    }
    #pragma unroll
    for (int j = 0; j < 4; ++j) {
        int n = nb + j * 16 + lm;
        #pragma unroll
        for (int v = 0; v < 8; ++v) {
            int m = m0 + v + half * 8;
            Y[(size_t)m * N + n] = acc[j][v] + bias[m];
        }
    }
}

// ---------------------------------------------------------------------------
// GroupNorm statistics (one block per group) -> stats[g] = {mean, rstd}
// ---------------------------------------------------------------------------
__global__ void cvd_gn_stats(const float* __restrict__ X, float* __restrict__ stats,
                             int Cg, int N) {
    __shared__ float s1[256];
    __shared__ float s2[256];
    int g = blockIdx.x;
    const float* base = X + (size_t)g * Cg * N;
    int total = Cg * N;
    float a = 0.f, b = 0.f;
    for (int i = threadIdx.x; i < total; i += blockDim.x) {
        float v = base[i];
        a += v; b += v * v;
    }
    s1[threadIdx.x] = a; s2[threadIdx.x] = b;
    __syncthreads();
    for (int off = 128; off > 0; off >>= 1) {
        if ((int)threadIdx.x < off) {
            s1[threadIdx.x] += s1[threadIdx.x + off];
            s2[threadIdx.x] += s2[threadIdx.x + off];
        }
        __syncthreads();
    }
    if (threadIdx.x == 0) {
        float inv = 1.f / (float)total;
        float mean = s1[0] * inv;
        float var  = s2[0] * inv - mean * mean;
        stats[g * 2 + 0] = mean;
        stats[g * 2 + 1] = rsqrtf(var + 1e-5f);
    }
}

// GN + exact GELU + residual (in-place on residual buffer)
__global__ void cvd_gn_gelu_res(const float* __restrict__ Xn, const float* __restrict__ stats,
                                const float* __restrict__ gamma, const float* __restrict__ beta,
                                float* __restrict__ XR, int Cn, int Cg, int N) {
    int idx = blockIdx.x * blockDim.x + threadIdx.x;
    int total = Cn * N;
    if (idx >= total) return;
    int c = idx / N;
    int g = c / Cg;
    float v = (Xn[idx] - stats[2 * g]) * stats[2 * g + 1] * gamma[c] + beta[c];
    XR[idx] = gelu_exact(v) + XR[idx];
}

// ---------------------------------------------------------------------------
// logits = dp - agg - dsout ; softmax over D=64 ; idl / unc / pmax
// ---------------------------------------------------------------------------
__global__ void cvd_softmax_stats(const float* __restrict__ dp, const float* __restrict__ agg,
                                  const float* __restrict__ dsout, float* __restrict__ idl,
                                  float* __restrict__ unc, float* __restrict__ pmax, int N) {
    int p = blockIdx.x * blockDim.x + threadIdx.x;
    if (p >= N) return;
    float maxl = -1e30f;
    for (int d = 0; d < D_; ++d) {
        float l = dp[d * N + p] - agg[d * N + p] - dsout[d * N + p];
        maxl = fmaxf(maxl, l);
    }
    float sum = 0.f, s1 = 0.f, s2 = 0.f;
    for (int d = 0; d < D_; ++d) {
        float l = dp[d * N + p] - agg[d * N + p] - dsout[d * N + p];
        float e = expf(l - maxl);
        float ib = 0.05f + (float)d * (1.95f / 63.f);
        sum += e; s1 += e * ib; s2 += e * ib * ib;
    }
    float id  = s1 / sum;
    float var = fmaxf(s2 / sum - id * id, 0.f);
    idl[p]  = id;
    unc[p]  = fmaxf(sqrtf(var), 1e-6f);
    pmax[p] = 1.f / sum;
}

// ---------------------------------------------------------------------------
// 3x3 conv 67->64 implicit-im2col WMMA GEMM. A pre-packed zero-padded f16
// (branch-free v8h loads). B gather is fully branch-free: clamped coords,
// cndmask-selected base pointer, unconditional load, masked select.
// ---------------------------------------------------------------------------
__device__ __forceinline__ float cvd_im2col_fetch(const float* __restrict__ ffr,
                                                  const float* __restrict__ imrr,
                                                  int k, int py, int px) {
    const int N = FH_ * FW_;
    int t = k / 67, ci = k % 67;
    int yy = py + t / 3 - 1;
    int xx = px + t % 3 - 1;
    bool inb = (yy >= 0) & (yy < FH_) & (xx >= 0) & (xx < FW_);
    int q = imin(imax(yy, 0), FH_ - 1) * FW_ + imin(imax(xx, 0), FW_ - 1);
    const float* base = (ci < C_) ? (ffr + (size_t)ci * N) : (imrr + (size_t)(ci - C_) * N);
    float v = base[q];                   // always executes from a safe address
    return inb ? v : 0.f;                // v_cndmask, no branch
}

__global__ void cvd_frconv_wmma(const float* __restrict__ ff, const float* __restrict__ imr,
                                const _Float16* __restrict__ Wp, const float* __restrict__ bias,
                                float* __restrict__ Y, int r) {
    const int N = FH_ * FW_;
    int wave = threadIdx.x >> 5;
    int lane = threadIdx.x & 31;
    int m0 = (wave & 3) * 16;
    int nb = blockIdx.x * 128 + (wave >> 2) * 64;
    int lm = lane & 15, half = lane >> 4;
    int py[4], px[4];
    #pragma unroll
    for (int j = 0; j < 4; ++j) {
        int n = nb + j * 16 + lm;
        py[j] = n / FW_; px[j] = n % FW_;
    }
    const float* ffr  = ff  + (size_t)r * C_ * N;
    const float* imrr = imr + (size_t)r * 3 * N;
    const _Float16* wrow = Wp + (size_t)(m0 + lm) * KFRP_;
    v8f acc[4] = {{}, {}, {}, {}};
    for (int kk = 0; kk < 576; kk += 32) {   // 18 full chunks
        __builtin_prefetch(wrow + kk + 32, 0, 3);
        v16h a = cat16(*reinterpret_cast<const v8h*>(wrow + kk + half * 8),
                       *reinterpret_cast<const v8h*>(wrow + kk + 16 + half * 8));
        #pragma unroll
        for (int j = 0; j < 4; ++j) {
            v16h b;
            #pragma unroll
            for (int e = 0; e < 16; ++e)
                b[e] = (_Float16)cvd_im2col_fetch(ffr, imrr, kk + half * 8 + (e & 7) + (e >> 3) * 16, py[j], px[j]);
            acc[j] = __builtin_amdgcn_wmma_f32_16x16x32_f16(false, a, false, b, (short)0,
                                                            acc[j], false, false);
        }
    }
    {   // partial chunk k in [576,608); A rows already zero-padded
        const int kk = 576;
        v16h a = cat16(*reinterpret_cast<const v8h*>(wrow + kk + half * 8),
                       *reinterpret_cast<const v8h*>(wrow + kk + 16 + half * 8));
        #pragma unroll
        for (int j = 0; j < 4; ++j) {
            v16h b;
            #pragma unroll
            for (int e = 0; e < 16; ++e) {
                int k = kk + half * 8 + (e & 7) + (e >> 3) * 16;
                float bv = cvd_im2col_fetch(ffr, imrr, imin(k, KFR_ - 1), py[j], px[j]);
                b[e] = (_Float16)((k < KFR_) ? bv : 0.f);
            }
            acc[j] = __builtin_amdgcn_wmma_f32_16x16x32_f16(false, a, false, b, (short)0,
                                                            acc[j], false, false);
        }
    }
    #pragma unroll
    for (int j = 0; j < 4; ++j) {
        int n = nb + j * 16 + lm;
        #pragma unroll
        for (int v = 0; v < 8; ++v) {
            int m = m0 + v + half * 8;
            float val = acc[j][v] + bias[m];
            Y[(size_t)m * N + n] = gelu_exact(val);
        }
    }
}

// ---------------------------------------------------------------------------
// output heads
// ---------------------------------------------------------------------------
__global__ void cvd_heads(const float* __restrict__ refined, const float* __restrict__ idf,
                          const float* __restrict__ uf, const float* __restrict__ pmf,
                          const float* __restrict__ res_w, const float* __restrict__ res_b,
                          const float* __restrict__ conf_w, const float* __restrict__ conf_b,
                          float* __restrict__ out, int r) {
    const int N = FH_ * FW_;
    int p = blockIdx.x * blockDim.x + threadIdx.x;
    if (p >= N) return;
    float sres = res_b[0], sconf = conf_b[0];
    for (int c = 0; c < C_; ++c) {
        float v = refined[(size_t)c * N + p];
        sres  += res_w[c]  * v;
        sconf += conf_w[c] * v;
    }
    const float interval = 1.95f / 63.f;
    float resv  = tanhf(sres) * interval;
    float inv_d = fminf(fmaxf(idf[p] + resv, 0.05f), 2.0f);
    float depth = 1.f / inv_d;
    float dunc  = fmaxf(uf[p] / fmaxf(inv_d * inv_d, 1e-6f), 0.01f);
    float sig   = 1.f / (1.f + expf(-sconf));
    float conf  = sig * pmf[p] / (1.f + uf[p] / interval);
    out[(size_t)r * N + p]            = depth;
    out[(size_t)(V_ + r) * N + p]     = dunc;
    out[(size_t)(2 * V_ + r) * N + p] = conf;
}

// ---------------------------------------------------------------------------
// host-side ds_block: dw3x3(f16 n-major out) -> WMMA pointwise -> GN -> +res
// ---------------------------------------------------------------------------
static void ds_block(float* X, _Float16* tAh, float* tB, const float* dw,
                     const _Float16* pwwh, const float* pwb, const float* g, const float* b,
                     float* stats, int H, int W, hipStream_t stream) {
    int N = H * W;                      // 5120 or 81920, both multiples of 128
    int total = C_ * N;
    cvd_dwconv3_f16t<<<(total + 255) / 256, 256, 0, stream>>>(X, dw, tAh, C_, H, W);
    cvd_pw_gemm_wmma<<<N / 128, 256, 0, stream>>>(tAh, pwwh, pwb, tB, N);
    cvd_gn_stats<<<8, 256, 0, stream>>>(tB, stats, C_ / 8, N);
    cvd_gn_gelu_res<<<(total + 255) / 256, 256, 0, stream>>>(tB, stats, g, b, X, C_, C_ / 8, N);
}

extern "C" void kernel_launch(void* const* d_in, const int* in_sizes, int n_in,
                              void* d_out, int out_size, void* d_ws, size_t ws_size,
                              hipStream_t stream) {
    (void)in_sizes; (void)n_in; (void)out_size; (void)ws_size;
    const float* lf   = (const float*)d_in[0];
    const float* ff   = (const float*)d_in[1];
    const float* imgs = (const float*)d_in[2];
    const float* Kin  = (const float*)d_in[3];
    const float* Ein  = (const float*)d_in[4];
    const float* dp_w = (const float*)d_in[5];
    const float* dp_b = (const float*)d_in[6];
    const float* blk[4][5];
    for (int i = 0; i < 4; ++i)
        for (int j = 0; j < 5; ++j) blk[i][j] = (const float*)d_in[7 + i * 5 + j];
    const float* fr_conv_w = (const float*)d_in[27];
    const float* fr_conv_b = (const float*)d_in[28];
    const float* res_w  = (const float*)d_in[29];
    const float* res_b  = (const float*)d_in[30];
    const float* conf_w = (const float*)d_in[31];
    const float* conf_b = (const float*)d_in[32];
    float* out = (float*)d_out;

    const int Nl = LH_ * LW_;          // 5120
    const int Nf = FH_ * FW_;          // 81920
    float* ws = (float*)d_ws;
    size_t off = 0;
    float* imgs_r = ws + off; off += (size_t)V_ * 3 * Nf;
    float* Ksb    = ws + off; off += 64;
    float* Kinvb  = ws + off; off += 64;
    float* w2cb   = ws + off; off += 64;
    float* stats  = ws + off; off += 64;
    float* aggb   = ws + off; off += (size_t)D_ * Nl;
    float* dpb    = ws + off; off += (size_t)D_ * Nl;
    float* idl    = ws + off; off += Nl;
    float* uncl   = ws + off; off += Nl;
    float* pml    = ws + off; off += Nl;
    float* idf    = ws + off; off += Nf;
    float* uff    = ws + off; off += Nf;
    float* pmf    = ws + off; off += Nf;
    float* Xb     = ws + off; off += (size_t)C_ * Nf;
    float* tB     = ws + off; off += (size_t)C_ * Nf;
    _Float16* tAh   = (_Float16*)(ws + off); off += (size_t)C_ * Nf / 2;      // n-major f16
    _Float16* lfhT  = (_Float16*)(ws + off); off += (size_t)V_ * C_ * Nl / 2; // n-major f16
    _Float16* dpwh  = (_Float16*)(ws + off); off += 4096 / 2;
    _Float16* pwwh[4];
    for (int i = 0; i < 4; ++i) { pwwh[i] = (_Float16*)(ws + off); off += 4096 / 2; }
    _Float16* frwh  = (_Float16*)(ws + off); off += (size_t)C_ * KFRP_ / 2;

    // one-time prep: matrices, image resize, f16 weight/activation packs
    cvd_setup_mats<<<1, 32, 0, stream>>>(Kin, Ein, Ksb, Kinvb, w2cb);
    cvd_resize_bilinear<<<(V_ * 3 * Nf + 255) / 256, 256, 0, stream>>>(
        imgs, imgs_r, V_ * 3, IH_, IW_, FH_, FW_);
    cvd_f32_to_f16<<<(4096 + 255) / 256, 256, 0, stream>>>(dp_w, dpwh, 4096);
    for (int i = 0; i < 4; ++i)
        cvd_f32_to_f16<<<(4096 + 255) / 256, 256, 0, stream>>>(blk[i][1], pwwh[i], 4096);
    cvd_pack_frw<<<(C_ * KFRP_ + 255) / 256, 256, 0, stream>>>(fr_conv_w, frwh);
    cvd_lf_to_f16t<<<(V_ * C_ * Nl + 255) / 256, 256, 0, stream>>>(lf, lfhT);

    for (int r = 0; r < V_; ++r) {
        // --- cost volume + lowres refinement (D=64 channels @ 64x80) ---
        cvd_cost_volume<<<(D_ * Nl + 255) / 256, 256, 0, stream>>>(
            lf, Ein, Ksb, Kinvb, w2cb, aggb, r);
        hipMemcpyAsync(Xb, aggb, sizeof(float) * (size_t)D_ * Nl,
                       hipMemcpyDeviceToDevice, stream);
        ds_block(Xb, tAh, tB, blk[0][0], pwwh[0], blk[0][2], blk[0][3], blk[0][4],
                 stats, LH_, LW_, stream);
        ds_block(Xb, tAh, tB, blk[1][0], pwwh[1], blk[1][2], blk[1][3], blk[1][4],
                 stats, LH_, LW_, stream);
        cvd_pw_gemm_wmma<<<Nl / 128, 256, 0, stream>>>(
            lfhT + (size_t)r * Nl * C_, dpwh, dp_b, dpb, Nl);
        cvd_softmax_stats<<<(Nl + 255) / 256, 256, 0, stream>>>(
            dpb, aggb, Xb, idl, uncl, pml, Nl);
        cvd_resize_bilinear<<<(Nf + 255) / 256, 256, 0, stream>>>(idl,  idf, 1, LH_, LW_, FH_, FW_);
        cvd_resize_bilinear<<<(Nf + 255) / 256, 256, 0, stream>>>(uncl, uff, 1, LH_, LW_, FH_, FW_);
        cvd_resize_bilinear<<<(Nf + 255) / 256, 256, 0, stream>>>(pml,  pmf, 1, LH_, LW_, FH_, FW_);
        // --- fullres refinement (C=64 channels @ 256x320) ---
        cvd_frconv_wmma<<<Nf / 128, 256, 0, stream>>>(ff, imgs_r, frwh, fr_conv_b, Xb, r);
        ds_block(Xb, tAh, tB, blk[2][0], pwwh[2], blk[2][2], blk[2][3], blk[2][4],
                 stats, FH_, FW_, stream);
        ds_block(Xb, tAh, tB, blk[3][0], pwwh[3], blk[3][2], blk[3][3], blk[3][4],
                 stats, FH_, FW_, stream);
        cvd_heads<<<(Nf + 255) / 256, 256, 0, stream>>>(
            Xb, idf, uff, pmf, res_w, res_b, conf_w, conf_b, out, r);
    }
}
